// HierarchicalCodebookGrounding_90752658964800
// MI455X (gfx1250) — matrix operand order repaired and checked by reading, not compiled, not run on hardware
//
#include <hip/hip_runtime.h>
#include <hip/hip_bf16.h>
#include <math.h>

typedef __bf16 bf16_t;
typedef __attribute__((ext_vector_type(16))) __bf16 v16bf;
typedef __attribute__((ext_vector_type(8)))  __bf16 v8bf;
typedef __attribute__((ext_vector_type(8)))  float  v8f;

#define D      320
#define NTOK   2048
#define NB     32
#define MTOT   1040
#define KCH    10      // D / 32

// ---------------- WMMA fragment helpers (16x16x32 bf16; contiguous k order used by this code) ----------------
// A (16x32): lanes 0-15 -> M=0..15, K base 0; lanes 16-31 -> same M, K base +16;
//            elems 0..15 -> K = kb+e
__device__ __forceinline__ v16bf load_a_frag(const bf16_t* base, int ld, int row0,
                                             int k0, int lane, int rows) {
  int m = row0 + (lane & 15);
  if (m >= rows) m = rows - 1;                 // clamped read for partial tiles
  int kb = k0 + ((lane & 16) ? 16 : 0);
  const bf16_t* p = base + (size_t)m * ld + kb;
  union { v16bf v; v8bf h[2]; } u;
  u.h[0] = *(const v8bf*)(p);
  u.h[1] = *(const v8bf*)(p + 8);
  return u.v;
}

// B (32x16) from "transposed" storage Bt[n][k] (rows = output col n, contiguous k):
// lane n = lane&15 ; lanes 16-31 hold K += 16 ; 16 contiguous bf16 = one 32B load
__device__ __forceinline__ v16bf load_b_frag(const bf16_t* bt, int ld, int n0,
                                             int k0, int lane) {
  int n  = n0 + (lane & 15);
  int kb = k0 + ((lane & 16) ? 16 : 0);
  return *(const v16bf*)(bt + (size_t)n * ld + kb);
}

__device__ __forceinline__ v8f wmma_bf16(v16bf a, v16bf b, v8f c) {
  return __builtin_amdgcn_wmma_f32_16x16x32_bf16(false, a, false, b, (short)0, c,
                                                 false, false);
}

// ---------------- weight transpose+convert, plain convert ----------------
__global__ void transcvt(const float* __restrict__ W, bf16_t* __restrict__ Wt,
                         int K, int N) {
  int i = blockIdx.x * 256 + threadIdx.x;
  if (i < K * N) {
    int n = i / K, k = i % K;
    Wt[i] = (bf16_t)W[k * N + n];   // Wt[n][k] = W[k][n]
  }
}

__global__ void cvtbf(const float* __restrict__ src, bf16_t* __restrict__ dst, int n) {
  int i = blockIdx.x * 256 + threadIdx.x;
  if (i < n) dst[i] = (bf16_t)src[i];
}

// ---------------- generic 16-row-strip GEMM, K=320, Ncols=320, 4 waves ----------------
// MODE 0: out bf16 = acc + bias[n]
// MODE 1: out bf16 = gelu(acc + bias2[batch(row)][n])          (h = gelu(X@W1a + g@W1b + b1))
// MODE 2: gate = sigmoid(acc + bias[n]); out bf16 = X + gate*g (A2)
// MODE 3: y = acc + bias[n]; LayerNorm over row -> fp32 out
// MODE 4: out fp32 = acc + bias[n]                             (gb1)
template<int MODE>
__global__ __launch_bounds__(128)
void gemm320(const bf16_t* __restrict__ A, const bf16_t* __restrict__ Wt, int rows,
             const float* __restrict__ bias, const float* __restrict__ bias2,
             const float* __restrict__ Xf, const float* __restrict__ grounded,
             const float* __restrict__ ln_g, const float* __restrict__ ln_b,
             bf16_t* __restrict__ outh, float* __restrict__ outf)
{
  __shared__ float ssum[16];
  __shared__ float ssq[16];

  int tid  = threadIdx.x;
  int lane = tid & 31;
  int wv   = tid >> 5;
  int r0   = blockIdx.x * 16;

  v8f acc[5] = {};
  for (int c = 0; c < KCH; ++c) {
    v16bf a = load_a_frag(A, D, r0, c * 32, lane, rows);
    for (int t = 0; t < 5; ++t) {
      v16bf bf = load_b_frag(Wt, D, (wv * 5 + t) * 16, c * 32, lane);
      acc[t] = wmma_bf16(a, bf, acc[t]);
    }
  }

  if (MODE == 3) {
    if (tid < 16) { ssum[tid] = 0.f; ssq[tid] = 0.f; }
    __syncthreads();
  }

  int mh = (lane & 16) ? 8 : 0;     // C layout: VGPR r -> row r (+8 for upper lanes)
  int nl = lane & 15;

  float vreg[5][8];
  for (int t = 0; t < 5; ++t) {
    int n = (wv * 5 + t) * 16 + nl;
    for (int r = 0; r < 8; ++r) {
      float v = acc[t][r];
      if (MODE == 1) {
        int m = r0 + r + mh;
        int b = m >> 11;                       // row / 2048
        v += bias2[b * D + n];
        v = 0.5f * v * (1.f + erff(v * 0.70710678118654752f));   // exact gelu
      } else {
        v += bias[n];
      }
      vreg[t][r] = v;
    }
  }

  if (MODE == 3) {
    for (int t = 0; t < 5; ++t)
      for (int r = 0; r < 8; ++r) {
        atomicAdd(&ssum[r + mh], vreg[t][r]);
        atomicAdd(&ssq[r + mh],  vreg[t][r] * vreg[t][r]);
      }
    __syncthreads();
  }

  for (int t = 0; t < 5; ++t) {
    int n = (wv * 5 + t) * 16 + nl;
    for (int r = 0; r < 8; ++r) {
      int m = r0 + r + mh;
      if (m >= rows) continue;
      float v = vreg[t][r];
      if (MODE == 0 || MODE == 1) {
        outh[(size_t)m * D + n] = (bf16_t)v;
      } else if (MODE == 2) {
        int b = m >> 11;
        float gate = 1.f / (1.f + __expf(-v));
        float g = grounded[b * D + n];
        float x = Xf[(size_t)m * D + n];
        outh[(size_t)m * D + n] = (bf16_t)(x + gate * g);
      } else if (MODE == 3) {
        float mu  = ssum[r + mh] * (1.f / 320.f);
        float var = ssq[r + mh] * (1.f / 320.f) - mu * mu;
        outf[(size_t)m * D + n] = (v - mu) * rsqrtf(var + 1e-5f) * ln_g[n] + ln_b[n];
      } else {
        outf[(size_t)m * D + n] = v;
      }
    }
  }
}

// ---------------- fused attention + row-norm: w_raw[b][m] = ||softmax(clip(Q.Kt^T/sqrt(D))) @ X|| / tau
// block = (m-tile of 16 codes) x (batch); 4 waves; full 16x2048 score matrix in LDS.
__global__ __launch_bounds__(128)
void attn_norm(const bf16_t* __restrict__ Qh, const bf16_t* __restrict__ Kth,
               const bf16_t* __restrict__ Xh, const float* __restrict__ log_tau,
               float* __restrict__ w_raw)
{
  extern __shared__ char smem[];
  float*  S    = (float*)smem;                                   // 16*2048 f32 = 128KB
  bf16_t* xs   = (bf16_t*)(smem + 16 * NTOK * 4);                // 32*320 bf16 = 20KB
  float*  red  = (float*)(smem + 16 * NTOK * 4 + 32 * D * 2);    // 128 f32
  float*  rowv = red + 128;                                      // 16 f32
  float*  hss  = rowv + 16;                                      // 16 f32

  int tid  = threadIdx.x;
  int lane = tid & 31;
  int wv   = tid >> 5;
  int mt   = blockIdx.x;
  int b    = blockIdx.y;

  const bf16_t* Q0 = Qh  + (size_t)mt * 16 * D;
  const bf16_t* Kb = Kth + (size_t)b * NTOK * D;
  const bf16_t* Xb = Xh  + (size_t)b * NTOK * D;

  int mh = (lane & 16) ? 8 : 0;
  int nl = lane & 15;

  // phase 1: S = clip(Q.Kt^T / sqrt(D))
  v16bf afr[KCH];
  for (int c = 0; c < KCH; ++c) afr[c] = load_a_frag(Q0, D, 0, c * 32, lane, 16);

  const float scl = 0.055901699437494745f;   // 1/sqrt(320)
  for (int nt = wv; nt < NTOK / 16; nt += 4) {
    v8f acc = {};
    for (int c = 0; c < KCH; ++c)
      acc = wmma_bf16(afr[c], load_b_frag(Kb, D, nt * 16, c * 32, lane), acc);
    for (int r = 0; r < 8; ++r) {
      float v = acc[r] * scl;
      v = fminf(fmaxf(v, -50.f), 50.f);
      S[(r + mh) * NTOK + nt * 16 + nl] = v;
    }
  }
  __syncthreads();

  // phase 1.5: row softmax (16 rows x 2048, 8 threads per row)
  {
    int row = tid >> 3, seg = tid & 7;
    float* Sr = S + row * NTOK;
    float lmax = -1e30f;
    for (int n = seg; n < NTOK; n += 8) lmax = fmaxf(lmax, Sr[n]);
    red[row * 8 + seg] = lmax;
    __syncthreads();
    if (seg == 0) {
      float m = red[row * 8];
      for (int j = 1; j < 8; ++j) m = fmaxf(m, red[row * 8 + j]);
      rowv[row] = m;
    }
    __syncthreads();
    float rmax = rowv[row], lsum = 0.f;
    for (int n = seg; n < NTOK; n += 8) {
      float e = __expf(Sr[n] - rmax);
      Sr[n] = e; lsum += e;
    }
    red[row * 8 + seg] = lsum;
    __syncthreads();
    if (seg == 0) {
      float s = 0.f;
      for (int j = 0; j < 8; ++j) s += red[row * 8 + j];
      rowv[row] = 1.f / s;
    }
    __syncthreads();
    float rinv = rowv[row];
    for (int n = seg; n < NTOK; n += 8) Sr[n] *= rinv;
  }

  if (tid < 16) hss[tid] = 0.f;

  // phase 2: H = P @ X, contraction over tokens in chunks of 32 (X staged in LDS)
  v8f acc2[5] = {};
  for (int nc = 0; nc < NTOK / 32; ++nc) {
    __syncthreads();
    for (int i = tid; i < (32 * D) / 8; i += 128) {
      int row = i / (D / 8);
      int c8  = i % (D / 8);
      *(v8bf*)(xs + row * D + c8 * 8) =
          *(const v8bf*)(Xb + (size_t)(nc * 32 + row) * D + c8 * 8);
    }
    __syncthreads();

    union { v16bf v; bf16_t e[16]; } pa;   // A frag of P from LDS (f32 -> bf16)
    {
      const float* Sm = S + nl * NTOK + nc * 32;
      for (int e = 0; e < 8; ++e) pa.e[e]     = (bf16_t)Sm[2 * mh + e];
      for (int e = 0; e < 8; ++e) pa.e[8 + e] = (bf16_t)Sm[2 * mh + 8 + e];
    }
    int kb2 = (lane & 16) ? 16 : 0;
    for (int t = 0; t < 5; ++t) {
      int d0 = (wv * 5 + t) * 16;
      union { v16bf v; bf16_t e[16]; } xb;  // B frag: B[k][n] = xs[k][d0+n]
      for (int e = 0; e < 16; ++e) xb.e[e] = xs[(kb2 + e) * D + d0 + nl];
      acc2[t] = wmma_bf16(pa.v, xb.v, acc2[t]);
    }
  }
  __syncthreads();

  for (int t = 0; t < 5; ++t)
    for (int r = 0; r < 8; ++r) {
      float v = acc2[t][r];
      atomicAdd(&hss[r + mh], v * v);
    }
  __syncthreads();

  if (tid < 16) {
    float tau = __expf(log_tau[0]) + 0.1f;
    tau = fminf(fmaxf(tau, 0.1f), 2.0f);
    w_raw[b * MTOT + mt * 16 + tid] = sqrtf(hss[tid]) / tau;
  }
}

// ---------------- per-batch softmax + top-k + hierarchical gating ----------------
__device__ void softmax_topk(float* buf, float* orig, int* flag,
                             float* rv, int* ri, float* sc,
                             int M, int K, int tid)
{
  float lmax = -1e30f;
  for (int i = tid; i < M; i += 256) lmax = fmaxf(lmax, buf[i]);
  rv[tid] = lmax; __syncthreads();
  for (int s = 128; s > 0; s >>= 1) {
    if (tid < s) rv[tid] = fmaxf(rv[tid], rv[tid + s]);
    __syncthreads();
  }
  float mx = rv[0]; __syncthreads();
  float lsum = 0.f;
  for (int i = tid; i < M; i += 256) { float e = __expf(buf[i] - mx); buf[i] = e; lsum += e; }
  rv[tid] = lsum; __syncthreads();
  for (int s = 128; s > 0; s >>= 1) { if (tid < s) rv[tid] += rv[tid + s]; __syncthreads(); }
  float inv = 1.f / rv[0]; __syncthreads();
  for (int i = tid; i < M; i += 256) { float p = buf[i] * inv; buf[i] = p; orig[i] = p; flag[i] = 0; }
  if (tid == 0) sc[0] = 0.f;
  __syncthreads();

  for (int it = 0; it < K; ++it) {
    float bv = -1e30f; int bi = 0x7fffffff;
    for (int i = tid; i < M; i += 256) {
      float v = buf[i];
      if (v > bv) { bv = v; bi = i; }
    }
    rv[tid] = bv; ri[tid] = bi; __syncthreads();
    for (int s = 128; s > 0; s >>= 1) {
      if (tid < s) {
        if (rv[tid + s] > rv[tid] ||
            (rv[tid + s] == rv[tid] && ri[tid + s] < ri[tid])) {
          rv[tid] = rv[tid + s]; ri[tid] = ri[tid + s];
        }
      }
      __syncthreads();
    }
    if (tid == 0) { int idx = ri[0]; flag[idx] = 1; sc[0] += orig[idx]; buf[idx] = -1e30f; }
    __syncthreads();
  }
  float inv2 = 1.f / (sc[0] + 1e-8f);
  for (int i = tid; i < M; i += 256) buf[i] = flag[i] ? orig[i] * inv2 : 0.f;
  __syncthreads();
}

__global__ __launch_bounds__(256)
void topk_gate(const float* __restrict__ w_raw, float* __restrict__ w_s)
{
  __shared__ float buf[800];
  __shared__ float orig[800];
  __shared__ int   flag[800];
  __shared__ float rv[256];
  __shared__ int   ri[256];
  __shared__ float sc[1];
  __shared__ float cat_s[20];
  __shared__ float type_s[200];

  int tid = threadIdx.x;
  int b   = blockIdx.x;
  const float* raw = w_raw + b * MTOT;
  float* out = w_s + b * MTOT;

  for (int i = tid; i < 20; i += 256) buf[i] = raw[i];
  __syncthreads();
  softmax_topk(buf, orig, flag, rv, ri, sc, 20, 5, tid);
  for (int i = tid; i < 20; i += 256) { out[i] = buf[i]; cat_s[i] = buf[i]; }
  __syncthreads();

  for (int i = tid; i < 200; i += 256) buf[i] = raw[20 + i] * cat_s[i / 10];
  __syncthreads();
  softmax_topk(buf, orig, flag, rv, ri, sc, 200, 10, tid);
  for (int i = tid; i < 200; i += 256) { out[20 + i] = buf[i]; type_s[i] = buf[i]; }
  __syncthreads();

  for (int i = tid; i < 800; i += 256) buf[i] = raw[220 + i] * type_s[i / 4];
  __syncthreads();
  softmax_topk(buf, orig, flag, rv, ri, sc, 800, 8, tid);
  for (int i = tid; i < 800; i += 256) out[220 + i] = buf[i];
  __syncthreads();

  for (int i = tid; i < 20; i += 256) buf[i] = raw[1020 + i];
  __syncthreads();
  softmax_topk(buf, orig, flag, rv, ri, sc, 20, 5, tid);
  for (int i = tid; i < 20; i += 256) out[1020 + i] = buf[i];
}

// ---------------- grounded reconstruction ----------------
__global__ __launch_bounds__(320)
void grounded_k(const float* __restrict__ w_s, const float* __restrict__ cat_codes,
                const float* __restrict__ type_codes, const float* __restrict__ var_codes,
                const float* __restrict__ spat_codes, const float* __restrict__ level_weights,
                float* __restrict__ grounded, bf16_t* __restrict__ groundedh)
{
  int b = blockIdx.x, d = threadIdx.x;
  float l[4];
  float m = fmaxf(fmaxf(level_weights[0], level_weights[1]),
                  fmaxf(level_weights[2], level_weights[3]));
  float s = 0.f;
  for (int i = 0; i < 4; ++i) { l[i] = __expf(level_weights[i] - m); s += l[i]; }
  for (int i = 0; i < 4; ++i) l[i] /= s;

  const float* ws = w_s + b * MTOT;
  float acc = 0.f;
  for (int c = 0; c < 20; ++c)  acc += l[0] * ws[c]        * cat_codes[c * D + d];
  for (int i = 0; i < 200; ++i) acc += l[1] * ws[20 + i]   * type_codes[i * D + d];
  for (int i = 0; i < 800; ++i) acc += l[2] * ws[220 + i]  * var_codes[i * D + d];
  for (int c = 0; c < 20; ++c)  acc += l[3] * ws[1020 + c] * spat_codes[c * D + d];
  grounded[b * D + d]  = acc;
  groundedh[b * D + d] = (bf16_t)acc;
}

// ---------------- launch ----------------
extern "C" void kernel_launch(void* const* d_in, const int* in_sizes, int n_in,
                              void* d_out, int out_size, void* d_ws, size_t ws_size,
                              hipStream_t stream)
{
  const float* X           = (const float*)d_in[0];
  const float* k_proj_w    = (const float*)d_in[1];
  const float* k_proj_b    = (const float*)d_in[2];
  const float* cat_codes   = (const float*)d_in[3];
  const float* type_codes  = (const float*)d_in[4];
  const float* var_codes   = (const float*)d_in[5];
  const float* spat_codes  = (const float*)d_in[6];
  const float* log_tau     = (const float*)d_in[7];
  const float* cat_proj_w  = (const float*)d_in[8];
  const float* cat_proj_b  = (const float*)d_in[9];
  const float* type_proj_w = (const float*)d_in[10];
  const float* type_proj_b = (const float*)d_in[11];
  const float* var_proj_w  = (const float*)d_in[12];
  const float* var_proj_b  = (const float*)d_in[13];
  const float* spat_proj_w = (const float*)d_in[14];
  const float* spat_proj_b = (const float*)d_in[15];
  const float* gate_w1     = (const float*)d_in[16];
  const float* gate_b1     = (const float*)d_in[17];
  const float* gate_w2     = (const float*)d_in[18];
  const float* gate_b2     = (const float*)d_in[19];
  const float* out_w       = (const float*)d_in[20];
  const float* out_b       = (const float*)d_in[21];
  const float* ln_g        = (const float*)d_in[22];
  const float* ln_b        = (const float*)d_in[23];
  const float* level_weights = (const float*)d_in[24];
  float* out = (float*)d_out;
  (void)in_sizes; (void)n_in; (void)out_size; (void)ws_size;

  char* wsb = (char*)d_ws;
  size_t off = 0;
  auto alloc = [&](size_t bytes) -> void* {
    void* p = wsb + off;
    off += (bytes + 255) & ~(size_t)255;
    return p;
  };
  const size_t BN = (size_t)NB * NTOK;                    // 65536 rows

  bf16_t* Xh  = (bf16_t*)alloc(BN * D * 2);
  bf16_t* Kth = (bf16_t*)alloc(BN * D * 2);               // later reused as A2
  bf16_t* Hh  = (bf16_t*)alloc(BN * D * 2);
  bf16_t* Wk  = (bf16_t*)alloc((size_t)D * D * 2);
  bf16_t* Wc  = (bf16_t*)alloc((size_t)D * D * 2);
  bf16_t* Wty = (bf16_t*)alloc((size_t)D * D * 2);
  bf16_t* Wv  = (bf16_t*)alloc((size_t)D * D * 2);
  bf16_t* Wsp = (bf16_t*)alloc((size_t)D * D * 2);
  bf16_t* W1a = (bf16_t*)alloc((size_t)D * D * 2);
  bf16_t* W1b = (bf16_t*)alloc((size_t)D * D * 2);
  bf16_t* W2  = (bf16_t*)alloc((size_t)D * D * 2);
  bf16_t* Wo  = (bf16_t*)alloc((size_t)D * D * 2);
  bf16_t* Ch  = (bf16_t*)alloc((size_t)MTOT * D * 2);
  bf16_t* Qh  = (bf16_t*)alloc((size_t)MTOT * D * 2);
  float*  w_raw = (float*)alloc((size_t)NB * MTOT * 4);
  float*  w_sp  = (float*)alloc((size_t)NB * MTOT * 4);
  float*  grnd  = (float*)alloc((size_t)NB * D * 4);
  bf16_t* grndh = (bf16_t*)alloc((size_t)NB * D * 2);
  float*  gb1   = (float*)alloc((size_t)NB * D * 4);

  int wg = (D * D + 255) / 256;
  transcvt<<<wg, 256, 0, stream>>>(k_proj_w,    Wk,  D, D);
  transcvt<<<wg, 256, 0, stream>>>(cat_proj_w,  Wc,  D, D);
  transcvt<<<wg, 256, 0, stream>>>(type_proj_w, Wty, D, D);
  transcvt<<<wg, 256, 0, stream>>>(var_proj_w,  Wv,  D, D);
  transcvt<<<wg, 256, 0, stream>>>(spat_proj_w, Wsp, D, D);
  transcvt<<<wg, 256, 0, stream>>>(gate_w1,         W1a, D, D);   // rows 0..319   (X part)
  transcvt<<<wg, 256, 0, stream>>>(gate_w1 + D * D, W1b, D, D);   // rows 320..639 (g part)
  transcvt<<<wg, 256, 0, stream>>>(gate_w2,     W2,  D, D);
  transcvt<<<wg, 256, 0, stream>>>(out_w,       Wo,  D, D);

  cvtbf<<<(int)((BN * D + 255) / 256), 256, 0, stream>>>(X, Xh, (int)(BN * D));
  cvtbf<<<(20 * D + 255) / 256, 256, 0, stream>>>(cat_codes,  Ch,             20 * D);
  cvtbf<<<(200 * D + 255) / 256, 256, 0, stream>>>(type_codes, Ch + 20 * D,   200 * D);
  cvtbf<<<(800 * D + 255) / 256, 256, 0, stream>>>(var_codes,  Ch + 220 * D,  800 * D);
  cvtbf<<<(20 * D + 255) / 256, 256, 0, stream>>>(spat_codes, Ch + 1020 * D,  20 * D);

  // Kt = X @ k_proj + b
  gemm320<0><<<(int)(BN / 16), 128, 0, stream>>>(Xh, Wk, (int)BN, k_proj_b,
      nullptr, nullptr, nullptr, nullptr, nullptr, Kth, nullptr);

  // Q projections (concatenated into Qh: cat | type | var | spat)
  gemm320<0><<<2, 128, 0, stream>>>(Ch,            Wc,  20,  cat_proj_b,
      nullptr, nullptr, nullptr, nullptr, nullptr, Qh,            nullptr);
  gemm320<0><<<13, 128, 0, stream>>>(Ch + 20 * D,  Wty, 200, type_proj_b,
      nullptr, nullptr, nullptr, nullptr, nullptr, Qh + 20 * D,   nullptr);
  gemm320<0><<<50, 128, 0, stream>>>(Ch + 220 * D, Wv,  800, var_proj_b,
      nullptr, nullptr, nullptr, nullptr, nullptr, Qh + 220 * D,  nullptr);
  gemm320<0><<<2, 128, 0, stream>>>(Ch + 1020 * D, Wsp, 20,  spat_proj_b,
      nullptr, nullptr, nullptr, nullptr, nullptr, Qh + 1020 * D, nullptr);

  // fused attention + norm (S matrix lives in 320KB LDS)
  size_t shb = (size_t)16 * NTOK * 4 + (size_t)32 * D * 2 + (128 + 16 + 16) * 4;
  attn_norm<<<dim3(MTOT / 16, NB), 128, shb, stream>>>(Qh, Kth, Xh, log_tau, w_raw);

  topk_gate<<<NB, 256, 0, stream>>>(w_raw, w_sp);
  grounded_k<<<NB, 320, 0, stream>>>(w_sp, cat_codes, type_codes, var_codes,
                                     spat_codes, level_weights, grnd, grndh);

  // gb1[b] = g[b] @ W1b + gate_b1
  gemm320<4><<<2, 128, 0, stream>>>(grndh, W1b, NB, gate_b1,
      nullptr, nullptr, nullptr, nullptr, nullptr, nullptr, gb1);

  // h = gelu(X @ W1a + gb1[batch])
  gemm320<1><<<(int)(BN / 16), 128, 0, stream>>>(Xh, W1a, (int)BN, nullptr,
      gb1, nullptr, nullptr, nullptr, nullptr, Hh, nullptr);

  // gate = sigmoid(h @ W2 + b2);  A2 = X + gate * g   (reuse Kth buffer)
  gemm320<2><<<(int)(BN / 16), 128, 0, stream>>>(Hh, W2, (int)BN, gate_b2,
      nullptr, X, grnd, nullptr, nullptr, Kth, nullptr);

  // out = LayerNorm(A2 @ out_w + out_b)
  gemm320<3><<<(int)(BN / 16), 128, 0, stream>>>(Kth, Wo, (int)BN, out_b,
      nullptr, nullptr, nullptr, ln_g, ln_b, nullptr, out);
}
